// VQVAE_22926535426160
// MI455X (gfx1250) — matrix operand order repaired and checked by reading, not compiled, not run on hardware
//
#include <hip/hip_runtime.h>

typedef __attribute__((ext_vector_type(16))) _Float16 v16h;
typedef __attribute__((ext_vector_type(8)))  float    v8f;

#define BN_EPS 1e-5f

// ---------------- conv1: 3->16, k4 s2 p1, 512^2 -> 256^2 (direct f32) -------
__global__ void k_conv1(const float* __restrict__ x, const float* __restrict__ w,
                        const float* __restrict__ b, float* __restrict__ y) {
    int o = blockIdx.x * 256 + threadIdx.x;                 // 33,554,432 outputs
    int ox = o & 255, oy = (o >> 8) & 255, c = (o >> 16) & 15, n = o >> 20;
    const float* xn = x + (long)n * 3 * 262144;
    const float* wc = w + c * 48;                           // (16,3,4,4)
    float acc = b[c];
    int iy0 = 2 * oy - 1, ix0 = 2 * ox - 1;
    #pragma unroll
    for (int ci = 0; ci < 3; ++ci) {
        const float* xp = xn + ci * 262144;
        #pragma unroll
        for (int ky = 0; ky < 4; ++ky) {
            int iy = iy0 + ky;
            int iyc = min(max(iy, 0), 511);
            bool vy = (unsigned)iy < 512u;
            #pragma unroll
            for (int kx = 0; kx < 4; ++kx) {
                int ix = ix0 + kx;
                int ixc = min(max(ix, 0), 511);
                float v = xp[iyc * 512 + ixc];              // unconditional load
                bool ok = vy && ((unsigned)ix < 512u);
                acc += ok ? wc[ci * 16 + ky * 4 + kx] * v : 0.f;
            }
        }
    }
    y[o] = acc;
}

// ---------------- per-channel batch stats (sum, sumsq) ----------------------
__global__ void k_stats(const float* __restrict__ y, float* __restrict__ s,
                        int C, int plane) {
    __shared__ float ssum[256], ssq[256];
    int tid = threadIdx.x;
    long f4 = (long)blockIdx.x * 256 + tid;                 // 1024 elems / block
    float4 v = ((const float4*)y)[f4];
    ssum[tid] = v.x + v.y + v.z + v.w;
    ssq[tid]  = v.x * v.x + v.y * v.y + v.z * v.z + v.w * v.w;
    __syncthreads();
    for (int off = 128; off > 0; off >>= 1) {
        if (tid < off) { ssum[tid] += ssum[tid + off]; ssq[tid] += ssq[tid + off]; }
        __syncthreads();
    }
    if (tid == 0) {
        int c = (int)((((long)blockIdx.x * 1024) / plane) % C);
        atomicAdd(&s[c], ssum[0]);
        atomicAdd(&s[C + c], ssq[0]);
    }
}

// ---------------- fused BatchNorm (batch stats) + ReLU, in place ------------
__global__ void k_bnrelu(float* __restrict__ y, const float* __restrict__ s,
                         const float* __restrict__ g, const float* __restrict__ be,
                         int C, int plane, float invN) {
    long i = (long)blockIdx.x * 256 + threadIdx.x;          // float4 index
    int c = (int)(((i * 4) / plane) % C);
    float m  = s[c] * invN;
    float vv = s[C + c] * invN - m * m;                     // biased variance
    float sc = g[c] * rsqrtf(vv + BN_EPS);
    float bb = be[c];
    float4 v = ((const float4*)y)[i];
    v.x = fmaxf(sc * (v.x - m) + bb, 0.f);
    v.y = fmaxf(sc * (v.y - m) + bb, 0.f);
    v.z = fmaxf(sc * (v.z - m) + bb, 0.f);
    v.w = fmaxf(sc * (v.w - m) + bb, 0.f);
    ((float4*)y)[i] = v;
}

// ---------------- conv2: 16->64, k4 s2 p1, 256^2 -> 128^2, WMMA f16 ---------
// One block = one output row (128 positions). Input rows staged in LDS as f16.
// GEMM: A = w2 [64 x 256] (K = ci*16 + ky*4 + kx), B = im2col [256 x 128].
__global__ void k_conv2_wmma(const float* __restrict__ h1, const float* __restrict__ w2,
                             const float* __restrict__ b2, float* __restrict__ y2) {
    __shared__ __align__(32) _Float16 Alds[16384];          // [mt][kt][lane][e] 32KB
    __shared__ _Float16 Bt[16 * 4 * 260];                   // [ci][ky][cc], cc = ix+1
    int tid = threadIdx.x;
    int n = blockIdx.x >> 7, oy = blockIdx.x & 127;         // 4096 blocks

    // A fragments in contiguous k order (k = kt*32 + 16*g + e), same order as B
    #pragma unroll
    for (int i = 0; i < 64; ++i) {
        int idx = tid * 64 + i;
        int e = idx & 15, lane = (idx >> 4) & 31, kt = (idx >> 9) & 7, mt = idx >> 12;
        int g = lane >> 4, m = mt * 16 + (lane & 15);
        int k = kt * 32 + g * 16 + e;
        Alds[idx] = (_Float16)w2[m * 256 + k];
    }
    // Input tile: 4 rows (iy = 2*oy-1+ky) x 16 ch, zero-padded halo (cc = ix+1)
    const float* inN = h1 + (long)n * 16 * 65536;
    for (int idx = tid; idx < 16 * 4 * 260; idx += 256) {
        int cc = idx % 260, rest = idx / 260;
        int ky = rest & 3, ci = rest >> 2;
        int iy = 2 * oy - 1 + ky, ix = cc - 1;
        float v = 0.f;
        if ((unsigned)iy < 256u && (unsigned)ix < 256u) v = inN[ci * 65536 + iy * 256 + ix];
        Bt[idx] = (_Float16)v;
    }
    __syncthreads();

    int wave = tid >> 5, lane = tid & 31, g = lane >> 4, ln = lane & 15;
    int ox = wave * 16 + ln;                                // full row per block
    v8f c0 = {0,0,0,0,0,0,0,0}, c1 = c0, c2 = c0, c3 = c0;
    const v16h* Af = (const v16h*)Alds;
    #pragma unroll
    for (int kt = 0; kt < 8; ++kt) {
        int ci = kt * 2 + g;                                // lane's 16-K slice == 4x4 patch
        const _Float16* bp = &Bt[ci * 1040 + 2 * ox];       // cc0 = 2*ox
        v16h bf;
        #pragma unroll
        for (int e = 0; e < 16; ++e)
            bf[e] = bp[(e >> 2) * 260 + (e & 3)];
        v16h a0 = Af[(0 * 8 + kt) * 32 + lane];
        v16h a1 = Af[(1 * 8 + kt) * 32 + lane];
        v16h a2 = Af[(2 * 8 + kt) * 32 + lane];
        v16h a3 = Af[(3 * 8 + kt) * 32 + lane];
        c0 = __builtin_amdgcn_wmma_f32_16x16x32_f16(false, a0, false, bf, (short)0, c0, false, false);
        c1 = __builtin_amdgcn_wmma_f32_16x16x32_f16(false, a1, false, bf, (short)0, c1, false, false);
        c2 = __builtin_amdgcn_wmma_f32_16x16x32_f16(false, a2, false, bf, (short)0, c2, false, false);
        c3 = __builtin_amdgcn_wmma_f32_16x16x32_f16(false, a3, false, bf, (short)0, c3, false, false);
    }
    long outBase = (long)n * 64 * 16384 + oy * 128 + ox;    // lane column N = ln
    #pragma unroll
    for (int r = 0; r < 8; ++r) {
        int co = g * 8 + r;                                 // C/D rows M = 8g + r
        y2[outBase + (long)(co +  0) * 16384] = c0[r] + b2[co];
        y2[outBase + (long)(co + 16) * 16384] = c1[r] + b2[co + 16];
        y2[outBase + (long)(co + 32) * 16384] = c2[r] + b2[co + 32];
        y2[outBase + (long)(co + 48) * 16384] = c3[r] + b2[co + 48];
    }
}

// -------- fused preq(64->2) + VQ(argmin of 3) + loss + postq(2->64) ---------
__global__ void k_vq(const float* __restrict__ h2, const float* __restrict__ pw,
                     const float* __restrict__ pb, const float* __restrict__ emb,
                     const float* __restrict__ qw, const float* __restrict__ qb,
                     float* __restrict__ dOut, float* __restrict__ lossA) {
    __shared__ float red[256];
    int tid = threadIdx.x;
    int p = blockIdx.x * 256 + tid;                         // 524,288 positions
    int n = p >> 14, hw = p & 16383;
    const float* h = h2 + (long)n * 64 * 16384 + hw;
    float q0 = pb[0], q1 = pb[1];
    #pragma unroll 8
    for (int ci = 0; ci < 64; ++ci) {
        float v = h[ci * 16384];
        q0 += pw[ci] * v;                                   // preq_w (2,64,1,1)
        q1 += pw[64 + ci] * v;
    }
    float best = 3.4e38f; int bi = 0;
    #pragma unroll
    for (int j = 0; j < 3; ++j) {                           // strict < == first-min
        float dx = q0 - emb[2 * j], dy = q1 - emb[2 * j + 1];
        float d = dx * dx + dy * dy;
        if (d < best) { best = d; bi = j; }
    }
    float e0 = emb[2 * bi], e1 = emb[2 * bi + 1];
    float dx = e0 - q0, dy = e1 - q1;
    red[tid] = dx * dx + dy * dy;
    __syncthreads();
    for (int off = 128; off > 0; off >>= 1) {
        if (tid < off) red[tid] += red[tid + off];
        __syncthreads();
    }
    if (tid == 0) atomicAdd(lossA, red[0]);
    float* dp = dOut + (long)n * 64 * 16384 + hw;           // postq_w (64,2,1,1)
    #pragma unroll 8
    for (int co = 0; co < 64; ++co)
        dp[co * 16384] = qw[2 * co] * e0 + qw[2 * co + 1] * e1 + qb[co];
}

// ------- convT1: 64->16, k4 s2 p1, 128^2 -> 256^2, parity-class WMMA --------
// Fixed output parity (oy&1, ox&1) -> exactly 2x2 taps: K = 64ch * 4taps = 256.
// One block = one output row; the 2 contributing input rows staged in LDS.
__global__ void k_convT1_wmma(const float* __restrict__ dIn, const float* __restrict__ w,
                              const float* __restrict__ b, float* __restrict__ y3) {
    __shared__ __align__(32) _Float16 Alds[8192];           // [px][kt][lane][e] 16KB
    __shared__ _Float16 Dt[64 * 2 * 132];                   // [ci][ty][cc], cc = ix+1
    int tid = threadIdx.x;
    int n = blockIdx.x >> 8, oy = blockIdx.x & 255;         // 8192 blocks
    int py = oy & 1, ky0 = 1 - py;
    int iyb = (oy + 1 - ky0) >> 1;                          // iy = iyb - ty

    #pragma unroll
    for (int i = 0; i < 32; ++i) {
        int idx = tid * 32 + i;
        int e = idx & 15, lane = (idx >> 4) & 31, kt = (idx >> 9) & 7, pxc = idx >> 12;
        int g = lane >> 4, m = lane & 15;
        int K = kt * 32 + g * 16 + e;
        int ci = K >> 2, t = K & 3;                         // k = ci*4 + (ty*2+tx)
        int kx0 = 1 - pxc;
        // dec_w1 (64,16,4,4): [in][out][ky][kx]
        Alds[idx] = (_Float16)w[ci * 256 + m * 16 + (ky0 + 2 * (t >> 1)) * 4 + (kx0 + 2 * (t & 1))];
    }
    const float* dN = dIn + (long)n * 64 * 16384;
    for (int idx = tid; idx < 64 * 2 * 132; idx += 256) {
        int cc = idx % 132, rest = idx / 132;
        int ty = rest & 1, ci = rest >> 1;
        int iy = iyb - ty, ix = cc - 1;
        float v = 0.f;
        if ((unsigned)iy < 128u && (unsigned)ix < 128u) v = dN[ci * 16384 + iy * 128 + ix];
        Dt[idx] = (_Float16)v;
    }
    __syncthreads();

    int wave = tid >> 5, lane = tid & 31, g = lane >> 4, ln = lane & 15;
    int oxb = wave * 32;
    long outN = (long)n * 16 * 65536 + oy * 256;
    const v16h* Af = (const v16h*)Alds;
    #pragma unroll
    for (int px = 0; px < 2; ++px) {
        int ox = oxb + px + 2 * ln;                         // 16 same-parity cols
        int ixb = (ox + 1 - (1 - px)) >> 1;                 // ix = ixb - tx
        v8f acc = {0,0,0,0,0,0,0,0};
        #pragma unroll
        for (int kt = 0; kt < 8; ++kt) {
            v16h bf;
            #pragma unroll
            for (int e = 0; e < 16; ++e) {
                int ci = kt * 8 + g * 4 + (e >> 2);
                int ty = (e >> 1) & 1, tx = e & 1;
                bf[e] = Dt[(ci * 2 + ty) * 132 + (ixb + 1 - tx)];
            }
            v16h a = Af[(px * 8 + kt) * 32 + lane];
            acc = __builtin_amdgcn_wmma_f32_16x16x32_f16(false, a, false, bf, (short)0, acc, false, false);
        }
        #pragma unroll
        for (int r = 0; r < 8; ++r) {
            int co = g * 8 + r;
            y3[outN + (long)co * 65536 + ox] = acc[r] + b[co];
        }
    }
}

// ---------------- convT2: 16->3, k4 s2 p1, 256^2 -> 512^2, + tanh -----------
__global__ void k_convT2_tanh(const float* __restrict__ h, const float* __restrict__ w,
                              const float* __restrict__ b, float* __restrict__ out) {
    int o = blockIdx.x * 256 + threadIdx.x;                 // 25,165,824 outputs
    int ox = o & 511, oy = (o >> 9) & 511;
    int c = (o >> 18) % 3;
    int n = o / 786432;
    const float* hn = h + (long)n * 16 * 65536;
    int ky0 = 1 - (oy & 1), kx0 = 1 - (ox & 1);
    int iyb = (oy + 1 - ky0) >> 1, ixb = (ox + 1 - kx0) >> 1;
    float acc = b[c];
    #pragma unroll
    for (int ty = 0; ty < 2; ++ty) {
        int iy = iyb - ty;
        int iyc = min(max(iy, 0), 255);
        #pragma unroll
        for (int tx = 0; tx < 2; ++tx) {
            int ix = ixb - tx;
            int ixc = min(max(ix, 0), 255);
            bool ok = ((unsigned)iy < 256u) && ((unsigned)ix < 256u);
            int ky = ky0 + 2 * ty, kx = kx0 + 2 * tx;
            float tap = 0.f;
            #pragma unroll
            for (int ci = 0; ci < 16; ++ci)                 // dec_w2 (16,3,4,4)
                tap += hn[ci * 65536 + iyc * 256 + ixc] * w[ci * 48 + c * 16 + ky * 4 + kx];
            acc += ok ? tap : 0.f;
        }
    }
    out[o] = tanhf(acc);
}

// ---------------- qloss = 1.2 * mean((e-q)^2) over N*2 elements -------------
__global__ void k_finalize(const float* __restrict__ loss, float* __restrict__ out) {
    *out = 1.2f * (*loss) / (524288.f * 2.f);
}

extern "C" void kernel_launch(void* const* d_in, const int* in_sizes, int n_in,
                              void* d_out, int out_size, void* d_ws, size_t ws_size,
                              hipStream_t stream) {
    (void)in_sizes; (void)n_in; (void)ws_size;
    const float* x    = (const float*)d_in[0];
    const float* ew1  = (const float*)d_in[1];
    const float* eb1  = (const float*)d_in[2];
    const float* eg1  = (const float*)d_in[3];
    const float* ebe1 = (const float*)d_in[4];
    const float* ew2  = (const float*)d_in[5];
    const float* eb2  = (const float*)d_in[6];
    const float* eg2  = (const float*)d_in[7];
    const float* ebe2 = (const float*)d_in[8];
    const float* pw   = (const float*)d_in[9];
    const float* pb   = (const float*)d_in[10];
    const float* emb  = (const float*)d_in[11];
    const float* qw   = (const float*)d_in[12];
    const float* qb   = (const float*)d_in[13];
    const float* dw1  = (const float*)d_in[14];
    const float* db1  = (const float*)d_in[15];
    const float* dg1  = (const float*)d_in[16];
    const float* dbe1 = (const float*)d_in[17];
    const float* dw2  = (const float*)d_in[18];
    const float* db2  = (const float*)d_in[19];

    char* ws = (char*)d_ws;
    float* bufA = (float*)ws;                                // 134,217,728 B
    float* bufB = (float*)(ws + 134217728L);                 // 134,217,728 B
    float* S    = (float*)(ws + 268435456L);                 // stats + loss
    float* S1 = S;            // 16ch: [sum x16][sq x16]
    float* S2 = S + 32;       // 64ch
    float* S3 = S + 160;      // 16ch
    float* lossA = S + 192;

    hipMemsetAsync(S, 0, 1024, stream);

    k_conv1      <<<131072, 256, 0, stream>>>(x, ew1, eb1, bufA);
    k_stats      <<<32768,  256, 0, stream>>>(bufA, S1, 16, 65536);
    k_bnrelu     <<<32768,  256, 0, stream>>>(bufA, S1, eg1, ebe1, 16, 65536, 1.f / 2097152.f);
    k_conv2_wmma <<<4096,   256, 0, stream>>>(bufA, ew2, eb2, bufB);
    k_stats      <<<32768,  256, 0, stream>>>(bufB, S2, 64, 16384);
    k_bnrelu     <<<32768,  256, 0, stream>>>(bufB, S2, eg2, ebe2, 64, 16384, 1.f / 524288.f);
    k_vq         <<<2048,   256, 0, stream>>>(bufB, pw, pb, emb, qw, qb, bufA, lossA);
    k_convT1_wmma<<<8192,   256, 0, stream>>>(bufA, dw1, db1, bufB);
    k_stats      <<<32768,  256, 0, stream>>>(bufB, S3, 16, 65536);
    k_bnrelu     <<<32768,  256, 0, stream>>>(bufB, S3, dg1, dbe1, 16, 65536, 1.f / 2097152.f);
    k_convT2_tanh<<<98304,  256, 0, stream>>>(bufB, dw2, db2, (float*)d_out);
    k_finalize   <<<1, 1,      0, stream>>>(lossA, (float*)d_out + (out_size - 1));
}
